// AttentionBlock_74268574482970
// MI455X (gfx1250) — matrix operand order repaired and checked by reading, compiled, not run on hardware
//
#include <hip/hip_runtime.h>

// ---------------------------------------------------------------------------
// Attention block for MI455X (gfx1250): all matmuls via v_wmma_f32_16x16x32_bf16
// Global->LDS staging via GLOBAL_LOAD_ASYNC_TO_LDS_B128 (ASYNCcnt) when available.
// ---------------------------------------------------------------------------

constexpr int BB = 4;      // batch
constexpr int SS = 2048;   // sequence
constexpr int DD = 1024;   // d_model
constexpr float EPS   = 1e-5f;
constexpr float SCALE = 0.03125f;  // 1/sqrt(1024)

typedef __attribute__((ext_vector_type(16))) __bf16 v16bf;
typedef __attribute__((ext_vector_type(8)))  float  v8f;

// exact parameter type of __builtin_amdgcn_global_load_async_to_lds_b128
// (from the round-2 diagnostic): generic pointer to vector_size(16) int
typedef int v4i_vs __attribute__((vector_size(4 * sizeof(int))));

union Frag { uint4 u4[2]; v16bf v; };

#if defined(__HIP_DEVICE_COMPILE__) && __has_builtin(__builtin_amdgcn_global_load_async_to_lds_b128)
#define USE_ASYNC_LDS 1
#else
#define USE_ASYNC_LDS 0
#endif

template <int N>
static __device__ __forceinline__ void waitAsync() {
#if defined(__HIP_DEVICE_COMPILE__) && __has_builtin(__builtin_amdgcn_s_wait_asynccnt)
  __builtin_amdgcn_s_wait_asynccnt(N);
#elif defined(__HIP_DEVICE_COMPILE__)
  asm volatile("s_wait_asynccnt %0" ::"n"(N) : "memory");
#endif
}

static __device__ __forceinline__ float waveSum(float v) {
  for (int m = 16; m; m >>= 1) v += __shfl_xor(v, m, 32);
  return v;
}
static __device__ __forceinline__ float waveMax(float v) {
  for (int m = 16; m; m >>= 1) v = fmaxf(v, __shfl_xor(v, m, 32));
  return v;
}

// ------------------------------ fp32 -> bf16 -------------------------------
__global__ void __launch_bounds__(256)
cvt_f32_bf16(const float* __restrict__ in, __bf16* __restrict__ out, int n) {
  int i = blockIdx.x * 256 + threadIdx.x;
  if (i < n) out[i] = (__bf16)in[i];
}

// ------------------------------ batched GEMM -------------------------------
// C = A[M,K] * Bt[N,K]^T (+bias), batched over blockIdx.z.
// mode 0: Cf fp32 [M,N] row-major
// mode 1: Cb bf16 [M,N] row-major
// mode 2: Cb bf16 [N,M] (transposed output)
__global__ void __launch_bounds__(256)
gemm_wmma_bf16(const __bf16* __restrict__ A, const __bf16* __restrict__ Bt,
               const float* __restrict__ bias,
               float* __restrict__ Cf, __bf16* __restrict__ Cb,
               int M, int N, int K,
               long long sA, long long sB, long long sC, int mode) {
  constexpr int BK  = 64;
  constexpr int LDT = 9;  // uint4 per LDS row: 64 bf16 = 8 uint4, +1 pad
  __shared__ uint4 As[2][128 * LDT];
  __shared__ uint4 Bs[2][128 * LDT];

  const int tid  = threadIdx.x;
  const int z    = blockIdx.z;
  const __bf16* Ab = A  + (long long)z * sA;
  const __bf16* Bb = Bt + (long long)z * sB;

  const int m0 = blockIdx.y * 128;
  const int n0 = blockIdx.x * 128;

  const int wave = tid >> 5;
  const int lane = tid & 31;
  const int half = lane >> 4;
  const int ln   = lane & 15;
  const int wm   = (wave >> 2) * 64;  // wave M offset (0 or 64)
  const int wn   = (wave & 3) * 32;   // wave N offset (0..96)

  v8f acc[4][2];
  const v8f vzero = {0.f, 0.f, 0.f, 0.f, 0.f, 0.f, 0.f, 0.f};
#pragma unroll
  for (int i = 0; i < 4; i++)
#pragma unroll
    for (int j = 0; j < 2; j++) acc[i][j] = vzero;

  // this thread's 4 tile slots (1024 uint4 per 128x64 bf16 tile)
  auto computeTile = [&](int buf) {
#pragma unroll
    for (int kd = 0; kd < 8; kd += 4) {  // kd in {0,4} -> K-chunk {0,32}
      Frag fb[2];
#pragma unroll
      for (int nt = 0; nt < 2; nt++) {
        const int rB = wn + nt * 16 + ln;
        fb[nt].u4[0] = Bs[buf][rB * LDT + kd + half * 2];
        fb[nt].u4[1] = Bs[buf][rB * LDT + kd + half * 2 + 1];
      }
#pragma unroll
      for (int mt = 0; mt < 4; mt++) {
        Frag fa;
        const int rA = wm + mt * 16 + ln;
        fa.u4[0] = As[buf][rA * LDT + kd + half * 2];
        fa.u4[1] = As[buf][rA * LDT + kd + half * 2 + 1];
#pragma unroll
        for (int nt = 0; nt < 2; nt++) {
          acc[mt][nt] = __builtin_amdgcn_wmma_f32_16x16x32_bf16(
              false, fa.v, false, fb[nt].v, (short)0, acc[mt][nt], false, false);
        }
      }
    }
  };

#if USE_ASYNC_LDS
  // Double-buffered pipeline: 8 async b128 copies per thread per tile,
  // direct global->LDS (no VGPR staging, no spills).
  auto issueTile = [&](int buf, int k0) {
#pragma unroll
    for (int i = 0; i < 4; i++) {
      const int slot = tid + i * 256;
      const int r = slot >> 3, c = slot & 7;
      __builtin_amdgcn_global_load_async_to_lds_b128(
          (v4i_vs*)(Ab + (long long)(m0 + r) * K + k0 + c * 8),
          (v4i_vs*)&As[buf][r * LDT + c], 0, 0);
      __builtin_amdgcn_global_load_async_to_lds_b128(
          (v4i_vs*)(Bb + (long long)(n0 + r) * K + k0 + c * 8),
          (v4i_vs*)&Bs[buf][r * LDT + c], 0, 0);
    }
  };

  issueTile(0, 0);
  int buf = 0;
  for (int k0 = 0; k0 < K; k0 += BK) {
    if (k0 + BK < K) {
      issueTile(buf ^ 1, k0 + BK);  // prefetch next tile into other buffer
      waitAsync<8>();               // own copies for CURRENT tile complete
    } else {
      waitAsync<0>();
    }
    __syncthreads();                // whole workgroup's copies complete
    computeTile(buf);
    __syncthreads();                // done reading before buffer is reused
    buf ^= 1;
  }
#else
  // Fallback: synchronous load -> immediate LDS store (no persistent
  // staging registers, avoids spills).
  for (int k0 = 0; k0 < K; k0 += BK) {
#pragma unroll
    for (int i = 0; i < 4; i++) {
      const int slot = tid + i * 256;
      const int r = slot >> 3, c = slot & 7;
      const uint4 a = *(const uint4*)(Ab + (long long)(m0 + r) * K + k0 + c * 8);
      const uint4 b = *(const uint4*)(Bb + (long long)(n0 + r) * K + k0 + c * 8);
      As[0][r * LDT + c] = a;
      Bs[0][r * LDT + c] = b;
    }
    __syncthreads();
    computeTile(0);
    __syncthreads();
  }
#endif

  // ------------------------------- epilogue --------------------------------
#pragma unroll
  for (int mt = 0; mt < 4; mt++)
#pragma unroll
    for (int nt = 0; nt < 2; nt++) {
      const int col = n0 + wn + nt * 16 + ln;
      const float bv = bias ? bias[col] : 0.f;
#pragma unroll
      for (int i = 0; i < 8; i++) {
        const int row = m0 + wm + mt * 16 + half * 8 + i;
        const float v = acc[mt][nt][i] + bv;
        if (mode == 0)
          Cf[(long long)z * sC + (long long)row * N + col] = v;
        else if (mode == 1)
          Cb[(long long)z * sC + (long long)row * N + col] = (__bf16)v;
        else
          Cb[(long long)z * sC + (long long)col * M + row] = (__bf16)v;
      }
    }
}

// ------------------------------ row softmax --------------------------------
__global__ void __launch_bounds__(256)
softmax_rows(const float* __restrict__ Sc, __bf16* __restrict__ P) {
  const long long row = blockIdx.x;  // 0 .. B*S-1
  const float* r  = Sc + row * SS;
  __bf16*      pr = P  + row * SS;
  const int tid = threadIdx.x, wave = tid >> 5, lane = tid & 31;
  __shared__ float red[8];

  float v[8], m = -1e30f;
#pragma unroll
  for (int j = 0; j < 8; j++) {
    v[j] = r[tid + j * 256] * SCALE;
    m = fmaxf(m, v[j]);
  }
  m = waveMax(m);
  if (lane == 0) red[wave] = m;
  __syncthreads();
  float bm = red[0];
#pragma unroll
  for (int i = 1; i < 8; i++) bm = fmaxf(bm, red[i]);
  __syncthreads();

  float s = 0.f;
#pragma unroll
  for (int j = 0; j < 8; j++) {
    v[j] = __expf(v[j] - bm);
    s += v[j];
  }
  s = waveSum(s);
  if (lane == 0) red[wave] = s;
  __syncthreads();
  float bs = 0.f;
#pragma unroll
  for (int i = 0; i < 8; i++) bs += red[i];
  const float inv = 1.f / bs;
#pragma unroll
  for (int j = 0; j < 8; j++) pr[tid + j * 256] = (__bf16)(v[j] * inv);
}

// -------------------- tril + residual + LayerNorm 1 ------------------------
__global__ void __launch_bounds__(256)
ln1_fused(const float* __restrict__ x, const float* __restrict__ attn,
          const float* __restrict__ g, const float* __restrict__ b,
          float* __restrict__ h1f, __bf16* __restrict__ h1b) {
  const long long row = blockIdx.x;
  const int srow = (int)(row & (SS - 1));  // s within batch
  const float* xr = x    + row * DD;
  const float* ar = attn + row * DD;
  const int tid = threadIdx.x, wave = tid >> 5, lane = tid & 31;
  __shared__ float r1[8], r2[8];

  float v[4]; float s = 0.f, q = 0.f;
#pragma unroll
  for (int j = 0; j < 4; j++) {
    const int d = tid * 4 + j;
    const float a = (d <= srow) ? ar[d] : 0.f;  // jnp.tril on [S,D] output
    v[j] = xr[d] + a;
    s += v[j]; q += v[j] * v[j];
  }
  s = waveSum(s); q = waveSum(q);
  if (lane == 0) { r1[wave] = s; r2[wave] = q; }
  __syncthreads();
  float ts = 0.f, tq = 0.f;
#pragma unroll
  for (int i = 0; i < 8; i++) { ts += r1[i]; tq += r2[i]; }
  const float mu = ts * (1.f / DD);
  const float var = tq * (1.f / DD) - mu * mu;
  const float rs = rsqrtf(var + EPS);
#pragma unroll
  for (int j = 0; j < 4; j++) {
    const int d = tid * 4 + j;
    const float y = (v[j] - mu) * rs * g[d] + b[d];
    h1f[row * DD + d] = y;
    h1b[row * DD + d] = (__bf16)y;
  }
}

// -------------------------- residual + LayerNorm 2 -------------------------
__global__ void __launch_bounds__(256)
ln2_fused(const float* __restrict__ h1, const float* __restrict__ lin,
          const float* __restrict__ g, const float* __restrict__ b,
          float* __restrict__ out) {
  const long long row = blockIdx.x;
  const float* hr = h1  + row * DD;
  const float* lr = lin + row * DD;
  const int tid = threadIdx.x, wave = tid >> 5, lane = tid & 31;
  __shared__ float r1[8], r2[8];

  float v[4]; float s = 0.f, q = 0.f;
#pragma unroll
  for (int j = 0; j < 4; j++) {
    const int d = tid * 4 + j;
    v[j] = hr[d] + lr[d];
    s += v[j]; q += v[j] * v[j];
  }
  s = waveSum(s); q = waveSum(q);
  if (lane == 0) { r1[wave] = s; r2[wave] = q; }
  __syncthreads();
  float ts = 0.f, tq = 0.f;
#pragma unroll
  for (int i = 0; i < 8; i++) { ts += r1[i]; tq += r2[i]; }
  const float mu = ts * (1.f / DD);
  const float var = tq * (1.f / DD) - mu * mu;
  const float rs = rsqrtf(var + EPS);
#pragma unroll
  for (int j = 0; j < 4; j++) {
    const int d = tid * 4 + j;
    out[row * DD + d] = (v[j] - mu) * rs * g[d] + b[d];
  }
}

// ---------------------------------------------------------------------------
extern "C" void kernel_launch(void* const* d_in, const int* in_sizes, int n_in,
                              void* d_out, int out_size, void* d_ws, size_t ws_size,
                              hipStream_t stream) {
  const float* x     = (const float*)d_in[0];
  const float* wq_w  = (const float*)d_in[1];
  const float* wq_b  = (const float*)d_in[2];
  const float* wk_w  = (const float*)d_in[3];
  const float* wk_b  = (const float*)d_in[4];
  const float* wv_w  = (const float*)d_in[5];
  const float* wv_b  = (const float*)d_in[6];
  const float* lin_w = (const float*)d_in[7];
  const float* lin_b = (const float*)d_in[8];
  const float* g1    = (const float*)d_in[9];
  const float* b1    = (const float*)d_in[10];
  const float* g2    = (const float*)d_in[11];
  const float* b2    = (const float*)d_in[12];
  float* out = (float*)d_out;

  const size_t nBSD = (size_t)BB * SS * DD;  // 8M
  const size_t nDDw = (size_t)DD * DD;       // 1M
  const size_t nBSS = (size_t)BB * SS * SS;  // 16M

  char* base = (char*)d_ws;
  size_t off = 0;
  auto take = [&](size_t bytes) -> char* {
    char* p = base + off;
    off += (bytes + 255) & ~(size_t)255;
    return p;
  };
  __bf16* xb   = (__bf16*)take(nBSD * 2);
  __bf16* wqb  = (__bf16*)take(nDDw * 2);
  __bf16* wkb  = (__bf16*)take(nDDw * 2);
  __bf16* wvb  = (__bf16*)take(nDDw * 2);
  __bf16* lwb  = (__bf16*)take(nDDw * 2);
  __bf16* qb   = (__bf16*)take(nBSD * 2);
  __bf16* kb   = (__bf16*)take(nBSD * 2);
  __bf16* vtb  = (__bf16*)take(nBSD * 2);  // V^T per batch: [D, S]
  float*  sc   = (float*) take(nBSS * 4);
  __bf16* pmat = (__bf16*)take(nBSS * 2);
  float*  attn = (float*) take(nBSD * 4);
  float*  h1f  = (float*) take(nBSD * 4);
  __bf16* h1b  = (__bf16*)take(nBSD * 2);
  float*  linf = sc;  // reuse score buffer for linear output

  const dim3 blk(256);
  cvt_f32_bf16<<<(int)((nBSD + 255) / 256), blk, 0, stream>>>(x, xb, (int)nBSD);
  cvt_f32_bf16<<<(int)((nDDw + 255) / 256), blk, 0, stream>>>(wq_w, wqb, (int)nDDw);
  cvt_f32_bf16<<<(int)((nDDw + 255) / 256), blk, 0, stream>>>(wk_w, wkb, (int)nDDw);
  cvt_f32_bf16<<<(int)((nDDw + 255) / 256), blk, 0, stream>>>(wv_w, wvb, (int)nDDw);
  cvt_f32_bf16<<<(int)((nDDw + 255) / 256), blk, 0, stream>>>(lin_w, lwb, (int)nDDw);

  const long long sSD  = (long long)SS * DD;
  const long long sSSx = (long long)SS * SS;
  const long long sDS  = (long long)DD * SS;

  // Q, K (bf16 row-major), V (bf16 transposed)
  dim3 gQKV(DD / 128, SS / 128, BB);
  gemm_wmma_bf16<<<gQKV, blk, 0, stream>>>(xb, wqb, wq_b, nullptr, qb,
                                           SS, DD, DD, sSD, 0, sSD, 1);
  gemm_wmma_bf16<<<gQKV, blk, 0, stream>>>(xb, wkb, wk_b, nullptr, kb,
                                           SS, DD, DD, sSD, 0, sSD, 1);
  gemm_wmma_bf16<<<gQKV, blk, 0, stream>>>(xb, wvb, wv_b, nullptr, vtb,
                                           SS, DD, DD, sSD, 0, sDS, 2);

  // scores = Q K^T (fp32)
  dim3 gSC(SS / 128, SS / 128, BB);
  gemm_wmma_bf16<<<gSC, blk, 0, stream>>>(qb, kb, nullptr, sc, nullptr,
                                          SS, SS, DD, sSD, sSD, sSSx, 0);

  // softmax rows -> P (bf16)
  softmax_rows<<<BB * SS, blk, 0, stream>>>(sc, pmat);

  // attn = P @ V (fp32); V columns contiguous via V^T
  dim3 gPV(DD / 128, SS / 128, BB);
  gemm_wmma_bf16<<<gPV, blk, 0, stream>>>(pmat, vtb, nullptr, attn, nullptr,
                                          SS, DD, SS, sSSx, sDS, sSD, 0);

  // h1 = LN1(x + tril(attn))
  ln1_fused<<<BB * SS, blk, 0, stream>>>(x, attn, g1, b1, h1f, h1b);

  // lin = h1 @ W_lin^T + b (fp32)
  gemm_wmma_bf16<<<gQKV, blk, 0, stream>>>(h1b, lwb, lin_b, linf, nullptr,
                                           SS, DD, DD, sSD, 0, sSD, 0);

  // out = LN2(h1 + lin)
  ln2_fused<<<BB * SS, blk, 0, stream>>>(h1f, linf, g2, b2, out);
}
